// AttnGNNLayer_26225070309675
// MI455X (gfx1250) — matrix operand order repaired and checked by reading, not compiled, not run on hardware
//
#include <hip/hip_runtime.h>
#include <hip/hip_bf16.h>
#include <math.h>

typedef __attribute__((ext_vector_type(16))) _Float16 v16h;
typedef __attribute__((ext_vector_type(8)))  _Float16 v8h;
typedef __attribute__((ext_vector_type(8)))  float    v8f;
typedef __attribute__((ext_vector_type(4)))  unsigned int v4u;
typedef __attribute__((ext_vector_type(8)))  int      v8i_;
typedef __attribute__((ext_vector_type(4)))  int      v4i_;

#define EPSV 1e-5f
#define NG 512   // B*M groups
#define NP 64    // points per group
#define NBR 16   // KNN

#if defined(__has_builtin)
#if __has_builtin(__builtin_amdgcn_tensor_load_to_lds) && \
    __has_builtin(__builtin_amdgcn_s_wait_tensorcnt)
#define HAVE_TDM 1
#endif
#endif

// ---------------- workspace layout (bytes) ----------------
static constexpr size_t OFF_IDX  = 0;                     // 512*64*16*4
static constexpr size_t OFF_XH   = 2097152;               // 512*64*64 f16
static constexpr size_t OFF_AE1  = OFF_XH  + 4194304;     // [128][64] f16
static constexpr size_t OFF_AE2  = OFF_AE1 + 16384;       // [256][64]
static constexpr size_t OFF_AC1  = OFF_AE2 + 32768;       // [64][192]
static constexpr size_t OFF_AC2  = OFF_AC1 + 24576;       // [192][64]
static constexpr size_t OFF_AX1  = OFF_AC2 + 24576;       // [256][192]
static constexpr size_t OFF_AX2  = OFF_AX1 + 98304;       // [512][256]
static constexpr size_t OFF_AR   = OFF_AX2 + 262144;      // [256][512]
static constexpr size_t OFF_AS1  = OFF_AR  + 262144;      // [256][256]
static constexpr size_t OFF_AS2  = OFF_AS1 + 131072;      // [256][256]
static constexpr size_t OFF_XCAT = OFF_AS2 + 131072;      // 512*64*192 f16
static constexpr size_t OFF_C1   = OFF_XCAT+ 12582912;    // 512*64*64 f16
static constexpr size_t OFF_XG   = OFF_C1  + 4194304;     // 512*64*192 f16
static constexpr size_t OFF_XE1  = OFF_XG  + 12582912;    // 512*64*256 f16
static constexpr size_t OFF_XMAX = OFF_XE1 + 16777216;    // 512*512 f16
static constexpr size_t OFF_XRF  = OFF_XMAX+ 524288;      // 2*256*256 f32
static constexpr size_t OFF_XRH  = OFF_XRF + 524288;      // 2*256*256 f16
static constexpr size_t OFF_H    = OFF_XRH + 262144;      // 2*256*256 f16

// ---------------- WMMA helpers ----------------
// A fragment per ISA 16-bit A 16x32 layout: lane m = lane&15, kh = (lane>>4)*8,
// elements 0..7 -> K = kh+i ; elements 8..15 -> K = kh+16+(i-8)
static __device__ __forceinline__ v16h frag_a(const _Float16* p) {
  v8h a0 = *(const v8h*)p;
  v8h a1 = *(const v8h*)(p + 16);
  return __builtin_shufflevector(a0, a1, 0,1,2,3,4,5,6,7,8,9,10,11,12,13,14,15);
}
// B fragment: lane n = lane&15, kb = (lane>>4)*8, elements 0..7 -> K = kb+i ; 8..15 -> K = kb+16+(i-8)
static __device__ __forceinline__ v16h frag_b(const _Float16* p) {
  v8h b0 = *(const v8h*)p;
  v8h b1 = *(const v8h*)(p + 16);
  return __builtin_shufflevector(b0, b1, 0,1,2,3,4,5,6,7,8,9,10,11,12,13,14,15);
}

// C += A[m0:16, :K] * B[:K, n0:16]; A row-major [M][lda], B stored [n][ldb] (K contiguous)
static __device__ __forceinline__ v8f gemm_tile(const _Float16* __restrict__ A, int lda,
                                                const _Float16* __restrict__ B, int ldb,
                                                int m0, int n0, int K) {
  const int lane = threadIdx.x & 31;
  const int am  = m0 + (lane & 15);
  const int akh = (lane >> 4) << 3;
  const int bn  = n0 + (lane & 15);
  const int bkb = (lane >> 4) << 3;
  v8f acc = {};
  for (int k0 = 0; k0 < K; k0 += 32) {
    v16h a = frag_a(A + (size_t)am * lda + k0 + akh);
    v16h b = frag_b(B + (size_t)bn * ldb + k0 + bkb);
    acc = __builtin_amdgcn_wmma_f32_16x16x32_f16(false, a, false, b, (short)0, acc,
                                                 false, false);
  }
  return acc;
}

// ---- Tensor Data Mover staging: copy rows x elems_per_row f16 tile ----------
// (global, row stride = row_stride elems) -> LDS packed [rows][elems_per_row].
static __device__ __forceinline__ void stage_b(_Float16* lds_dst,
                                               const _Float16* gsrc,
                                               int elems_per_row, int rows,
                                               int row_stride) {
#ifdef HAVE_TDM
  if (threadIdx.x < 32) {
    unsigned long long ga = (unsigned long long)(size_t)gsrc;
    unsigned lds = (unsigned)(size_t)(void*)lds_dst;
    v4u g0;
    g0[0] = 1u;                                    // count=1 (valid user D#)
    g0[1] = lds;                                   // lds_addr (bytes)
    g0[2] = (unsigned)(ga & 0xffffffffu);          // global_addr[31:0]
    g0[3] = (unsigned)((ga >> 32) & 0x01ffffffu)   // global_addr[56:32]
            | (2u << 30);                          // type=2 ("image")
    const unsigned td0 = (unsigned)row_stride;     // tensor_dim0 (elements)
    const unsigned nr  = (unsigned)rows;
    v8i_ g1;
    g1[0] = (int)(1u << 16);                       // wg_mask=0, data_size=1 (2B)
    g1[1] = (int)((td0 & 0xffffu) << 16);          // ab_addr=0 | tensor_dim0 lo16
    g1[2] = (int)((td0 >> 16) | (nr << 16));       // tensor_dim0 hi | tensor_dim1 lo
    g1[3] = (int)(((unsigned)elems_per_row) << 16);// tensor_dim1 hi=0 | tile_dim0
    g1[4] = (int)(nr & 0xffffu);                   // tile_dim1=rows, tile_dim2=0
    g1[5] = (int)td0;                              // tensor_dim0_stride lo32
    g1[6] = 0;                                     // stride hi16 | dim1_stride lo
    g1[7] = 0;
    v4i_ z4 = {0, 0, 0, 0};
    v8i_ z8 = {0, 0, 0, 0, 0, 0, 0, 0};
    __builtin_amdgcn_tensor_load_to_lds(g0, g1, z4, z4, z8, 0);
    __builtin_amdgcn_s_wait_tensorcnt(0);
  }
#else
  for (int e = threadIdx.x; e < rows * elems_per_row; e += blockDim.x) {
    int r = e / elems_per_row, c = e - r * elems_per_row;
    lds_dst[e] = gsrc[(size_t)r * row_stride + c];
  }
#endif
}

// ---------------- kernel 0: weights -> f16 (fold EdgeConv [Wa; Wb-Wa]) --------
__global__ void k_prep(const float* __restrict__ e1w, const float* __restrict__ e2w,
                       const float* __restrict__ c1w, const float* __restrict__ c2w,
                       const float* __restrict__ x1w, const float* __restrict__ x2w,
                       const float* __restrict__ rw,  const float* __restrict__ s1w,
                       const float* __restrict__ s2w,
                       _Float16* Ae1, _Float16* Ae2, _Float16* Ac1, _Float16* Ac2,
                       _Float16* Ax1, _Float16* Ax2, _Float16* Ar,  _Float16* As1,
                       _Float16* As2) {
  const int tid = blockIdx.x * blockDim.x + threadIdx.x;
  const int nt  = gridDim.x * blockDim.x;
  for (int i = tid; i < 128 * 64; i += nt) {
    int r = i >> 6, c = i & 63;
    float v = (r < 64) ? e1w[r * 128 + c]
                       : (e1w[(r - 64) * 128 + 64 + c] - e1w[(r - 64) * 128 + c]);
    Ae1[i] = (_Float16)v;
  }
  for (int i = tid; i < 256 * 64; i += nt) {
    int r = i >> 6, c = i & 63;
    float v = (r < 128) ? e2w[r * 128 + c]
                        : (e2w[(r - 128) * 128 + 64 + c] - e2w[(r - 128) * 128 + c]);
    Ae2[i] = (_Float16)v;
  }
  for (int i = tid; i < 64 * 192;  i += nt) Ac1[i] = (_Float16)c1w[i];
  for (int i = tid; i < 192 * 64;  i += nt) Ac2[i] = (_Float16)c2w[i];
  for (int i = tid; i < 256 * 192; i += nt) Ax1[i] = (_Float16)x1w[i];
  for (int i = tid; i < 512 * 256; i += nt) Ax2[i] = (_Float16)x2w[i];
  for (int i = tid; i < 256 * 512; i += nt) Ar[i]  = (_Float16)rw[i];
  for (int i = tid; i < 256 * 256; i += nt) As1[i] = (_Float16)s1w[i];
  for (int i = tid; i < 256 * 256; i += nt) As2[i] = (_Float16)s2w[i];
}

// ---------------- kernel 1: per-group KNN + build f16 activations -------------
__global__ void k_knn(const float* __restrict__ xyz, const float* __restrict__ feats,
                      int* __restrict__ idx, _Float16* __restrict__ xh) {
  const int g = blockIdx.x;   // 0..511
  const int p = threadIdx.x;  // 0..63
  __shared__ float px[NP][3];
  __shared__ float xx[NP];
  __shared__ float pd[NP][NP];
  const float x0 = xyz[((size_t)g * NP + p) * 3 + 0];
  const float x1 = xyz[((size_t)g * NP + p) * 3 + 1];
  const float x2 = xyz[((size_t)g * NP + p) * 3 + 2];
  px[p][0] = x0; px[p][1] = x1; px[p][2] = x2;
  xx[p] = x0 * x0 + x1 * x1 + x2 * x2;
  // x = concat(xyz, feats): 64 channels, f16, layout [g][n][c]
  _Float16* xr = xh + (size_t)g * NP * 64 + (size_t)p * 64;
  xr[0] = (_Float16)x0; xr[1] = (_Float16)x1; xr[2] = (_Float16)x2;
  const float* fr = feats + ((size_t)g * NP + p) * 61;
  for (int c = 0; c < 61; ++c) xr[3 + c] = (_Float16)fr[c];
  __syncthreads();
  for (int m = 0; m < NP; ++m) {
    float inner = x0 * px[m][0] + x1 * px[m][1] + x2 * px[m][2];
    pd[p][m] = 2.f * inner - xx[p] - xx[m];
  }
  unsigned long long taken = 0ull;
  int* op = idx + ((size_t)g * NP + p) * NBR;
  for (int k = 0; k < NBR; ++k) {
    float best = -3.4e38f; int bi = 0;
    for (int m = 0; m < NP; ++m) {
      if ((taken >> m) & 1ull) continue;
      float v = pd[p][m];
      if (v > best) { best = v; bi = m; }
    }
    taken |= 1ull << bi;
    op[k] = bi;
  }
}

// -------- kernel 2: fused EdgeConv: R=[Wa;Wb-Wa]@x, gather+bn+relu+max_k ------
__global__ void k_edge(const _Float16* __restrict__ A,   // [2*O][64] f16
                       const _Float16* __restrict__ xin, int ldin,
                       const int* __restrict__ idx,
                       const float* __restrict__ gam, const float* __restrict__ bet,
                       _Float16* __restrict__ out,       // xcat + coff, stride 192
                       int O) {
  const int g = blockIdx.x;
  const int tid = threadIdx.x;                 // 256 threads / 8 waves
  __shared__ __align__(16) _Float16 Bsh[NP * 64];
  __shared__ float Rs[256 * NP];               // up to 2*128 rows x 64 cols f32
  __shared__ int idxs[NP * NBR];
  __builtin_prefetch(A, 0, 0);
  // 2-D strided tile via TDM (EdgeConv2 reads 64 of 192 channels per row)
  stage_b(Bsh, xin + (size_t)g * NP * ldin, 64, NP, ldin);
  for (int e = tid; e < NP * NBR; e += 256) idxs[e] = idx[(size_t)g * NP * NBR + e];
  __syncthreads();
  const int R = 2 * O;
  const int T = (R >> 4) << 2;                 // m-tiles * 4 n-tiles
  const int wave = tid >> 5, lane = tid & 31;
  for (int t = wave; t < T; t += 8) {
    int m0 = (t >> 2) << 4, n0 = (t & 3) << 4;
    v8f acc = gemm_tile(A, 64, Bsh, 64, m0, n0, 64);
    int n  = n0 + (lane & 15);
    int mb = m0 + ((lane >> 4) << 3);
#pragma unroll
    for (int r = 0; r < 8; ++r) Rs[(mb + r) * NP + n] = acc[r];
  }
  __syncthreads();
  const float s = rsqrtf(1.f + EPSV);
  for (int e = tid; e < O * NP; e += 256) {
    int o = e >> 6, n = e & 63;
    float q  = Rs[(O + o) * NP + n];
    float gs = gam[o] * s, bb = bet[o];
    const float* Pr = &Rs[o * NP];
    float best = 0.f;                          // relu outputs are >= 0
    for (int k = 0; k < NBR; ++k) {
      float y = gs * (Pr[idxs[n * NBR + k]] + q) + bb;
      y = y > 0.f ? y : 0.f;
      best = y > best ? y : best;
    }
    out[(size_t)g * NP * 192 + (size_t)n * 192 + o] = (_Float16)best;
  }
}

// -------- kernel 3: per-group GEMM + bn + relu (cal1 / exp1) ------------------
__global__ void k_gemm_rbn(const _Float16* __restrict__ A,
                           const _Float16* __restrict__ Bg,   // [512][64][Kdim] f16
                           const float* __restrict__ gam, const float* __restrict__ bet,
                           _Float16* __restrict__ out, int ldo, int O, int Kdim) {
  const int g = blockIdx.x;
  const int tid = threadIdx.x;
  __shared__ __align__(16) _Float16 Bsh[NP * 256];
  __builtin_prefetch(A, 0, 0);
  stage_b(Bsh, Bg + (size_t)g * NP * Kdim, Kdim, NP, Kdim);
  __syncthreads();
  const float s = rsqrtf(1.f + EPSV);
  const int wave = tid >> 5, lane = tid & 31, nw = blockDim.x >> 5;
  const int T = (O >> 4) << 2;
  for (int t = wave; t < T; t += nw) {
    int m0 = (t >> 2) << 4, n0 = (t & 3) << 4;
    v8f acc = gemm_tile(A, Kdim, Bsh, Kdim, m0, n0, Kdim);
    int n  = n0 + (lane & 15);
    int mb = m0 + ((lane >> 4) << 3);
#pragma unroll
    for (int r = 0; r < 8; ++r) {
      int o = mb + r;
      float y = gam[o] * s * acc[r] + bet[o];
      y = y > 0.f ? y : 0.f;
      out[(size_t)g * NP * ldo + (size_t)n * ldo + o] = (_Float16)y;
    }
  }
}

// -------- kernel 4: cal2 + sigmoid gate: xg = sigmoid(cal2@c1 + bias) * xcat --
__global__ void k_cal2gate(const _Float16* __restrict__ A,    // [192][64]
                           const _Float16* __restrict__ c1,   // [512][64][64]
                           const float* __restrict__ cbias,
                           const _Float16* __restrict__ xcat, // [512][64][192]
                           _Float16* __restrict__ xg) {
  const int g = blockIdx.x;
  const int tid = threadIdx.x;
  __shared__ __align__(16) _Float16 Bsh[NP * 64];
  __builtin_prefetch(A, 0, 0);
  stage_b(Bsh, c1 + (size_t)g * NP * 64, 64, NP, 64);
  __syncthreads();
  const int wave = tid >> 5, lane = tid & 31;
  for (int t = wave; t < 48; t += 8) {          // 12 m-tiles x 4 n-tiles
    int m0 = (t >> 2) << 4, n0 = (t & 3) << 4;
    v8f acc = gemm_tile(A, 64, Bsh, 64, m0, n0, 64);
    int n  = n0 + (lane & 15);
    int mb = m0 + ((lane >> 4) << 3);
#pragma unroll
    for (int r = 0; r < 8; ++r) {
      int o = mb + r;
      float v  = acc[r] + cbias[o];
      float sg = 1.f / (1.f + __expf(-v));
      size_t ix = (size_t)g * NP * 192 + (size_t)n * 192 + o;
      xg[ix] = (_Float16)(sg * (float)xcat[ix]);
    }
  }
}

// -------- kernel 5: exp2 + bn + relu + max over points (in-register) ----------
__global__ void k_exp2max(const _Float16* __restrict__ A,    // [512][256]
                          const _Float16* __restrict__ xe1,  // [512][64][256]
                          const float* __restrict__ gam, const float* __restrict__ bet,
                          _Float16* __restrict__ xmax) {     // [512][512]
  const int g = blockIdx.x;
  const int tid = threadIdx.x;
  __shared__ __align__(16) _Float16 Bsh[NP * 256];
  __builtin_prefetch(A, 0, 0);
  stage_b(Bsh, xe1 + (size_t)g * NP * 256, 256, NP, 256);
  __syncthreads();
  const float s = rsqrtf(1.f + EPSV);
  const int wave = tid >> 5, lane = tid & 31;
  for (int mt = wave; mt < 32; mt += 8) {
    int m0 = mt << 4;
    float cm[8];
#pragma unroll
    for (int r = 0; r < 8; ++r) cm[r] = 0.f;    // relu outputs >= 0
    for (int nt = 0; nt < 4; ++nt) {
      v8f acc = gemm_tile(A, 256, Bsh, 256, m0, nt << 4, 256);
#pragma unroll
      for (int r = 0; r < 8; ++r) {
        int o = m0 + r + ((lane >> 4) << 3);
        float y = gam[o] * s * acc[r] + bet[o];
        y = y > 0.f ? y : 0.f;
        cm[r] = y > cm[r] ? y : cm[r];
      }
    }
#pragma unroll
    for (int mask = 1; mask < 16; mask <<= 1)
#pragma unroll
      for (int r = 0; r < 8; ++r)
        cm[r] = fmaxf(cm[r], __shfl_xor(cm[r], mask, 32));
    if ((lane & 15) == 0) {
      int mb = m0 + ((lane >> 4) << 3);
#pragma unroll
      for (int r = 0; r < 8; ++r) xmax[(size_t)g * 512 + mb + r] = (_Float16)cm[r];
    }
  }
}

// -------- kernel 6: red GEMM + bn + relu + (x+x) bn -> xr (f32 + f16) ---------
__global__ void k_red(const _Float16* __restrict__ A,      // [256][512]
                      const _Float16* __restrict__ xmax,   // [512][512]
                      const float* __restrict__ rg, const float* __restrict__ rb,
                      const float* __restrict__ n1g, const float* __restrict__ n1b,
                      float* __restrict__ xrf, _Float16* __restrict__ xrh) {
  const int b = blockIdx.x;                     // 0..1
  const _Float16* Bg = xmax + (size_t)b * 256 * 512;   // rows m, ld 512
  const int tid = threadIdx.x, wave = tid >> 5, lane = tid & 31;
  const float s = rsqrtf(1.f + EPSV);
  __builtin_prefetch(A, 0, 0);
  for (int t = wave; t < 256; t += 8) {         // 16 x 16 tiles, K = 512
    int m0 = (t >> 4) << 4, n0 = (t & 15) << 4;
    v8f acc = gemm_tile(A, 512, Bg, 512, m0, n0, 512);
    int n  = n0 + (lane & 15);                  // point m
    int mb = m0 + ((lane >> 4) << 3);
#pragma unroll
    for (int r = 0; r < 8; ++r) {
      int o = mb + r;
      float y = rg[o] * s * acc[r] + rb[o];
      y = y > 0.f ? y : 0.f;
      float v = n1g[o] * s * (y + y) + n1b[o];
      size_t ix = ((size_t)(b * 256 + n)) * 256 + o;
      xrf[ix] = v;
      xrh[ix] = (_Float16)v;
    }
  }
}

// -------- kernel 7: sc1 GEMM + bias + relu -> h (f16) -------------------------
__global__ void k_sc1(const _Float16* __restrict__ A,      // [256][256]
                      const _Float16* __restrict__ xrh,    // [2][256][256]
                      const float* __restrict__ s1b, _Float16* __restrict__ hbuf) {
  const int b = blockIdx.x;
  const _Float16* Bg = xrh + (size_t)b * 256 * 256;
  const int tid = threadIdx.x, wave = tid >> 5, lane = tid & 31;
  __builtin_prefetch(A, 0, 0);
  for (int t = wave; t < 256; t += 8) {
    int m0 = (t >> 4) << 4, n0 = (t & 15) << 4;
    v8f acc = gemm_tile(A, 256, Bg, 256, m0, n0, 256);
    int n  = n0 + (lane & 15);
    int mb = m0 + ((lane >> 4) << 3);
#pragma unroll
    for (int r = 0; r < 8; ++r) {
      int o = mb + r;
      float y = acc[r] + s1b[o];
      y = y > 0.f ? y : 0.f;
      hbuf[((size_t)(b * 256 + n)) * 256 + o] = (_Float16)y;
    }
  }
}

// -------- kernel 8: sc2 GEMM + bias, residual, final bn -> d_out (f32) --------
__global__ void k_sc2(const _Float16* __restrict__ A,      // [256][256]
                      const _Float16* __restrict__ hbuf,   // [2][256][256]
                      const float* __restrict__ s2b,
                      const float* __restrict__ xrf,
                      const float* __restrict__ n2g, const float* __restrict__ n2b,
                      float* __restrict__ outp) {          // (B, 256, M)
  const int b = blockIdx.x;
  const _Float16* Bg = hbuf + (size_t)b * 256 * 256;
  const int tid = threadIdx.x, wave = tid >> 5, lane = tid & 31;
  const float s = rsqrtf(1.f + EPSV);
  __builtin_prefetch(A, 0, 0);
  for (int t = wave; t < 256; t += 8) {
    int m0 = (t >> 4) << 4, n0 = (t & 15) << 4;
    v8f acc = gemm_tile(A, 256, Bg, 256, m0, n0, 256);
    int n  = n0 + (lane & 15);                  // point m
    int mb = m0 + ((lane >> 4) << 3);
#pragma unroll
    for (int r = 0; r < 8; ++r) {
      int o = mb + r;
      float hv = acc[r] + s2b[o];
      float xv = xrf[((size_t)(b * 256 + n)) * 256 + o];
      float v  = n2g[o] * s * (xv + hv) + n2b[o];
      outp[(size_t)b * 256 * 256 + (size_t)o * 256 + n] = v;
    }
  }
}

// ---------------- host launcher ----------------
extern "C" void kernel_launch(void* const* d_in, const int* in_sizes, int n_in,
                              void* d_out, int out_size, void* d_ws, size_t ws_size,
                              hipStream_t stream) {
  (void)in_sizes; (void)n_in; (void)out_size; (void)ws_size;
  const float* xyz   = (const float*)d_in[0];
  const float* feats = (const float*)d_in[1];
  const float* e1w = (const float*)d_in[2];
  const float* e1g = (const float*)d_in[3];
  const float* e1b = (const float*)d_in[4];
  const float* e2w = (const float*)d_in[5];
  const float* e2g = (const float*)d_in[6];
  const float* e2b = (const float*)d_in[7];
  const float* c1w = (const float*)d_in[8];
  const float* c1g = (const float*)d_in[9];
  const float* c1b = (const float*)d_in[10];
  const float* c2w   = (const float*)d_in[11];
  const float* c2bias= (const float*)d_in[12];
  const float* x1w = (const float*)d_in[13];
  const float* x1g = (const float*)d_in[14];
  const float* x1b = (const float*)d_in[15];
  const float* x2w = (const float*)d_in[16];
  const float* x2g = (const float*)d_in[17];
  const float* x2b = (const float*)d_in[18];
  const float* rw  = (const float*)d_in[19];
  const float* rg  = (const float*)d_in[20];
  const float* rb  = (const float*)d_in[21];
  const float* s1w = (const float*)d_in[22];
  const float* s1b = (const float*)d_in[23];
  const float* s2w = (const float*)d_in[24];
  const float* s2b = (const float*)d_in[25];
  const float* n1g = (const float*)d_in[26];
  const float* n1b = (const float*)d_in[27];
  const float* n2g = (const float*)d_in[28];
  const float* n2b = (const float*)d_in[29];

  char* ws = (char*)d_ws;
  int*      idx  = (int*)     (ws + OFF_IDX);
  _Float16* xh   = (_Float16*)(ws + OFF_XH);
  _Float16* Ae1  = (_Float16*)(ws + OFF_AE1);
  _Float16* Ae2  = (_Float16*)(ws + OFF_AE2);
  _Float16* Ac1  = (_Float16*)(ws + OFF_AC1);
  _Float16* Ac2  = (_Float16*)(ws + OFF_AC2);
  _Float16* Ax1  = (_Float16*)(ws + OFF_AX1);
  _Float16* Ax2  = (_Float16*)(ws + OFF_AX2);
  _Float16* Ar   = (_Float16*)(ws + OFF_AR);
  _Float16* As1  = (_Float16*)(ws + OFF_AS1);
  _Float16* As2  = (_Float16*)(ws + OFF_AS2);
  _Float16* xcat = (_Float16*)(ws + OFF_XCAT);
  _Float16* c1bf = (_Float16*)(ws + OFF_C1);
  _Float16* xg   = (_Float16*)(ws + OFF_XG);
  _Float16* xe1  = (_Float16*)(ws + OFF_XE1);
  _Float16* xmax = (_Float16*)(ws + OFF_XMAX);
  float*    xrf  = (float*)   (ws + OFF_XRF);
  _Float16* xrh  = (_Float16*)(ws + OFF_XRH);
  _Float16* hbuf = (_Float16*)(ws + OFF_H);
  float*    outp = (float*)d_out;

  k_prep<<<64, 256, 0, stream>>>(e1w, e2w, c1w, c2w, x1w, x2w, rw, s1w, s2w,
                                 Ae1, Ae2, Ac1, Ac2, Ax1, Ax2, Ar, As1, As2);
  k_knn<<<NG, NP, 0, stream>>>(xyz, feats, idx, xh);
  // EdgeConv 1: C=64 -> O=64, writes xcat[:, :, 0:64]
  k_edge<<<NG, 256, 0, stream>>>(Ae1, xh, 64, idx, e1g, e1b, xcat, 64);
  // EdgeConv 2: reads xcat[:, :, 0:64], writes xcat[:, :, 64:192]
  k_edge<<<NG, 256, 0, stream>>>(Ae2, xcat, 192, idx, e2g, e2b, xcat + 64, 128);
  // cal1: 64 x 192 GEMM + bn + relu
  k_gemm_rbn<<<NG, 256, 0, stream>>>(Ac1, xcat, c1g, c1b, c1bf, 64, 64, 192);
  // cal2 + sigmoid gate
  k_cal2gate<<<NG, 256, 0, stream>>>(Ac2, c1bf, c2bias, xcat, xg);
  // exp1: 256 x 192 GEMM + bn + relu
  k_gemm_rbn<<<NG, 256, 0, stream>>>(Ax1, xg, x1g, x1b, xe1, 256, 256, 192);
  // exp2 + bn + relu + max over points
  k_exp2max<<<NG, 256, 0, stream>>>(Ax2, xe1, x2g, x2b, xmax);
  // red + bn + relu + (x+x) bn
  k_red<<<2, 256, 0, stream>>>(Ar, xmax, rg, rb, n1g, n1b, xrf, xrh);
  // sc1 + bias + relu
  k_sc1<<<2, 256, 0, stream>>>(As1, xrh, s1b, hbuf);
  // sc2 + bias + residual + final bn -> output
  k_sc2<<<2, 256, 0, stream>>>(As2, hbuf, s2b, xrf, n2g, n2b, outp);
}
